// SchTrack_16990890623099
// MI455X (gfx1250) — matrix operand order repaired and checked by reading, not compiled, not run on hardware
//
#include <hip/hip_runtime.h>
#include <hip/hip_bf16.h>
#include <stdint.h>

typedef __attribute__((ext_vector_type(16))) __bf16 v16bf;
typedef __attribute__((ext_vector_type(8)))  float  v8f;

#define HWC 16384   // H*W
#define WW  128
#define CC  64
#define PP  8513    // (C+2)*C + C*C + C + C + C + 1
#define OFF_W1 4224
#define OFF_W2 8320
#define OFF_B0 8384
#define OFF_B1 8448
#define OFF_B2 8512

union BfPack { uint32_t u[8]; v16bf v; };

// ---------------- Stage 1: transpose features (B,C,HW) f32 -> (B,HW,C) bf16 ----------------
__global__ __launch_bounds__(256) void st_prep_feat(const float* __restrict__ sch,
                                                    __bf16* __restrict__ featT) {
    int idx = blockIdx.x * 256 + threadIdx.x;     // over B*C*HW = 4,194,304
    int pix = idx & (HWC - 1);
    int t   = idx >> 14;
    int c   = t & (CC - 1);
    int b   = t >> 6;
    featT[((size_t)b * HWC + pix) * CC + c] = (__bf16)sch[idx];
}

// ---------------- Stage 2: per-instance MLP via WMMA bf16 ----------------
__global__ __launch_bounds__(256, 2) void st_mlp(const float* __restrict__ convw,
                                                 const int*   __restrict__ pre_ind,
                                                 const __bf16* __restrict__ featT,
                                                 float* __restrict__ logits) {
    __shared__ __bf16 w0s[64 * 64];
    __shared__ __bf16 w1s[64 * 64];
    __shared__ alignas(16) float w2s[64], b0s[64], b1s[64], w0xs[64], w0ys[64];
    __shared__ float  b2s;

    const int n   = blockIdx.y;            // instance 0..63
    const int b   = n >> 4;                // batch
    const int tid = threadIdx.x;
    const float* params = convw + (size_t)n * PP;

    // Stage weights into LDS as bf16 (w0 channel part skips the 2 rel columns)
    for (int i = tid; i < 4096; i += 256) {
        int o = i >> 6, c = i & 63;
        w0s[i] = (__bf16)params[o * 66 + 2 + c];
        w1s[i] = (__bf16)params[OFF_W1 + i];
    }
    if (tid < 64) {
        w2s[tid]  = params[OFF_W2 + tid];
        b0s[tid]  = params[OFF_B0 + tid];
        b1s[tid]  = params[OFF_B1 + tid];
        w0xs[tid] = params[tid * 66 + 0];
        w0ys[tid] = params[tid * 66 + 1];
    }
    if (tid == 0) b2s = params[OFF_B2];
    __syncthreads();

    const int lane = tid & 31;
    const int wave = tid >> 5;
    const int hi   = lane >> 4;            // lane half (ISA operand layouts)
    const int ln   = lane & 15;

    const int   ind = pre_ind[n];
    const float xs  = (float)(ind & (WW - 1));
    const float ys  = (float)(ind >> 7);

    const int colWave = blockIdx.x * 256 + wave * 32;   // 32 pixels per wave

    // A-operand loader: 16x32 A tile in contiguous k order, same as B (lane<16: K0..15; lane>=16: K16..31)
    auto loadA = [&](const __bf16* wlds, int t, int kk) -> v16bf {
        BfPack A;
        const __bf16* base = wlds + (t * 16 + ln) * 64 + kk * 32 + hi * 16;
        const uint32_t* pa = (const uint32_t*)base;
        const uint32_t* pb = (const uint32_t*)(base + 8);
#pragma unroll
        for (int j = 0; j < 4; ++j) { A.u[j] = pa[j]; A.u[4 + j] = pb[j]; }
        return A.v;
    };

#pragma unroll 1
    for (int sub = 0; sub < 2; ++sub) {
        const int col = colWave + sub * 16 + ln;        // this lane's pixel
        // ---- B operand (features): K contiguous per pixel; lanes<16: K0..15, lanes>=16: K16..31
        const __bf16* fcol = featT + ((size_t)b * HWC + col) * CC;
        BfPack Bk0, Bk1;
        {
            const uint32_t* p0 = (const uint32_t*)(fcol + hi * 16);
            const uint32_t* p1 = (const uint32_t*)(fcol + 32 + hi * 16);
#pragma unroll
            for (int j = 0; j < 8; ++j) { Bk0.u[j] = p0[j]; Bk1.u[j] = p1[j]; }
        }

        // ---- GEMM0: H0(64xN) = W0ch(64x64) * X(64xN)
        v8f acc[4];
#pragma unroll
        for (int t = 0; t < 4; ++t)
#pragma unroll
            for (int r = 0; r < 8; ++r) acc[t][r] = 0.f;
#pragma unroll
        for (int t = 0; t < 4; ++t) {
            acc[t] = __builtin_amdgcn_wmma_f32_16x16x32_bf16(false, loadA(w0s, t, 0),
                                                             false, Bk0.v, (short)0, acc[t], false, false);
            acc[t] = __builtin_amdgcn_wmma_f32_16x16x32_bf16(false, loadA(w0s, t, 1),
                                                             false, Bk1.v, (short)0, acc[t], false, false);
        }

        // ---- epilogue0: fold rel channels + bias, ReLU (float4 LDS reads, not hoisted tables)
        const float gx = (float)(col & (WW - 1));
        const float gy = (float)(col >> 7);
        const float rx = (xs - gx) * 0.0078125f;
        const float ry = (ys - gy) * 0.0078125f;
#pragma unroll
        for (int t = 0; t < 4; ++t) {
            const int ob = t * 16 + hi * 8;
            float bv[8], xv[8], yv[8];
            *(float4*)&bv[0] = ((const float4*)(b0s  + ob))[0];
            *(float4*)&bv[4] = ((const float4*)(b0s  + ob))[1];
            *(float4*)&xv[0] = ((const float4*)(w0xs + ob))[0];
            *(float4*)&xv[4] = ((const float4*)(w0xs + ob))[1];
            *(float4*)&yv[0] = ((const float4*)(w0ys + ob))[0];
            *(float4*)&yv[4] = ((const float4*)(w0ys + ob))[1];
#pragma unroll
            for (int r = 0; r < 8; ++r) {
                float v = acc[t][r] + bv[r] + xv[r] * rx + yv[r] * ry;
                acc[t][r] = fmaxf(v, 0.f);
            }
        }

        // ---- build BOTH GEMM1 B-operands first so acc dies before acc2 is live
        v16bf bb[2];
#pragma unroll
        for (int kk = 0; kk < 2; ++kk) {
            v8f aA = acc[2 * kk], aB = acc[2 * kk + 1];
#pragma unroll
            for (int r = 0; r < 8; ++r) {
                float oa = __shfl_xor(aA[r], 16, 32);
                float ob = __shfl_xor(aB[r], 16, 32);
                bb[kk][r]     = (__bf16)(hi ? ob    : aA[r]);   // K = kk*32 + hi*16 + r
                bb[kk][8 + r] = (__bf16)(hi ? aB[r] : oa);      // K = kk*32 + hi*16 + 8 + r
            }
        }

        // ---- GEMM1
        v8f acc2[4];
#pragma unroll
        for (int t = 0; t < 4; ++t)
#pragma unroll
            for (int r = 0; r < 8; ++r) acc2[t][r] = 0.f;
#pragma unroll
        for (int kk = 0; kk < 2; ++kk)
#pragma unroll
            for (int t = 0; t < 4; ++t)
                acc2[t] = __builtin_amdgcn_wmma_f32_16x16x32_bf16(false, loadA(w1s, t, kk),
                                                                  false, bb[kk], (short)0, acc2[t], false, false);

        // ---- epilogue1 + layer2 dot product
        float part = 0.f;
#pragma unroll
        for (int t = 0; t < 4; ++t) {
            const int ob = t * 16 + hi * 8;
            float bv[8], wv[8];
            *(float4*)&bv[0] = ((const float4*)(b1s + ob))[0];
            *(float4*)&bv[4] = ((const float4*)(b1s + ob))[1];
            *(float4*)&wv[0] = ((const float4*)(w2s + ob))[0];
            *(float4*)&wv[4] = ((const float4*)(w2s + ob))[1];
#pragma unroll
            for (int r = 0; r < 8; ++r) {
                float v = fmaxf(acc2[t][r] + bv[r], 0.f);
                part = fmaf(wv[r], v, part);
            }
        }
        part += __shfl_xor(part, 16, 32);
        if (hi == 0) logits[(size_t)n * HWC + col] = part + b2s;
    }
}

// ---------------- Stage 3: 3x3 NMS (keep value iff it equals window max) ----------------
__global__ __launch_bounds__(256) void st_nms(const float* __restrict__ raw,
                                              float* __restrict__ hm) {
    int idx = blockIdx.x * 256 + threadIdx.x;      // over 64*16384
    int pix = idx & (HWC - 1);
    int nm  = idx >> 14;
    int x = pix & (WW - 1), y = pix >> 7;
    const float* base = raw + (size_t)nm * HWC;
    float v = base[pix];
    float mx = v;
#pragma unroll
    for (int dy = -1; dy <= 1; ++dy)
#pragma unroll
        for (int dx = -1; dx <= 1; ++dx) {
            int yy = y + dy, xx = x + dx;
            if (yy >= 0 && yy < WW && xx >= 0 && xx < WW)
                mx = fmaxf(mx, base[yy * WW + xx]);
        }
    hm[idx] = (v == mx) ? v : 0.f;
}

// ---------------- Stage 4: top-32 via monotonic-key threshold scan ----------------
__global__ __launch_bounds__(256) void st_topk(const float* __restrict__ hm,
                                               float* __restrict__ scores,
                                               int*   __restrict__ inds,
                                               float* __restrict__ ys0,
                                               float* __restrict__ xs0) {
    __shared__ unsigned long long wbest[8];
    __shared__ unsigned long long sel;
    const int nm  = blockIdx.x;
    const int tid = threadIdx.x;
    const float* src = hm + (size_t)nm * HWC;

    unsigned long long keys[64];
#pragma unroll
    for (int s = 0; s < 64; ++s) {
        int idx = s * 256 + tid;                 // coalesced
        uint32_t u = __float_as_uint(src[idx]);
        uint32_t k = (u & 0x80000000u) ? ~u : (u | 0x80000000u);   // order-preserving
        keys[s] = ((unsigned long long)k << 32) | (unsigned long long)(0x3FFFu - idx);
    }

    unsigned long long prev = ~0ull;
    for (int kk = 0; kk < 32; ++kk) {
        unsigned long long best = 0ull;
#pragma unroll
        for (int s = 0; s < 64; ++s) {
            unsigned long long key = keys[s];
            if (key < prev && key > best) best = key;
        }
#pragma unroll
        for (int off = 16; off > 0; off >>= 1) {
            unsigned long long o = __shfl_down(best, off, 32);
            if (o > best) best = o;
        }
        if ((tid & 31) == 0) wbest[tid >> 5] = best;
        __syncthreads();
        if (tid == 0) {
            unsigned long long bb = wbest[0];
            for (int wv = 1; wv < 8; ++wv) if (wbest[wv] > bb) bb = wbest[wv];
            sel = bb;
            uint32_t k = (uint32_t)(bb >> 32);
            uint32_t u = (k & 0x80000000u) ? (k & 0x7FFFFFFFu) : ~k;
            int idx = 0x3FFF - (int)(bb & 0xFFFFu);
            int pos = nm * 32 + kk;
            scores[pos] = __uint_as_float(u);
            inds[pos]   = idx;
            ys0[pos]    = (float)(idx >> 7);
            xs0[pos]    = (float)(idx & (WW - 1));
        }
        __syncthreads();
        prev = sel;
    }
}

extern "C" void kernel_launch(void* const* d_in, const int* in_sizes, int n_in,
                              void* d_out, int out_size, void* d_ws, size_t ws_size,
                              hipStream_t stream) {
    const float* sch     = (const float*)d_in[0];
    const float* convw   = (const float*)d_in[1];
    const int*   pre_ind = (const int*)d_in[2];
    // d_in[3] = nms_kernel (3), d_in[4] = track_K (32): compile-time constants here

    __bf16* featT   = (__bf16*)d_ws;                                   // 8 MB
    float*  rawLog  = (float*)((char*)d_ws + (size_t)8 * 1024 * 1024); // 4 MB

    float* out    = (float*)d_out;
    float* scores = out;                 // 2048
    int*   inds   = (int*)(out + 2048);  // 2048 (int32 bits)
    float* ysO    = out + 4096;          // 2048
    float* xsO    = out + 6144;          // 2048
    float* hm     = out + 8192;          // 1,048,576

    st_prep_feat<<<16384, 256, 0, stream>>>(sch, featT);
    dim3 g2(64, 64);                                   // pixel tiles x instances
    st_mlp<<<g2, 256, 0, stream>>>(convw, pre_ind, featT, rawLog);
    st_nms<<<4096, 256, 0, stream>>>(rawLog, hm);
    st_topk<<<64, 256, 0, stream>>>(hm, scores, inds, ysO, xsO);
}
